// SCA_20736102105415
// MI455X (gfx1250) — matrix operand order repaired and checked by reading, not compiled, not run on hardware
//
#include <hip/hip_runtime.h>

#define DIMC 768
#define HEADS 12
#define HEAD_DIM 64
#define SEQ 1024
#define BATCH 8
#define M_TOT (BATCH * SEQ)     // 8192
#define QKV_N (3 * DIMC)        // 2304
#define NBH (BATCH * HEADS)     // 96
#define ATT_SCALE 0.125f        // 64^-0.5

typedef __attribute__((ext_vector_type(16))) _Float16 v16h;
typedef __attribute__((ext_vector_type(8)))  _Float16 v8h;
typedef __attribute__((ext_vector_type(4)))  _Float16 v4h;
typedef __attribute__((ext_vector_type(8)))  float    v8f;
typedef __attribute__((ext_vector_type(4)))  float    v4f;

// ---- WMMA fragment loaders (wave32 layouts per CDNA5 ISA 7.12.2) ----

// A (16x32 f16): lane L holds row M=L%16; halves 0-7 = K kb..kb+7,
// halves 8-15 = K kb+16..kb+23, kb = 8*(L/16).
static __device__ __forceinline__ v16h frag_a_f16(const _Float16* __restrict__ A,
                                                  int ld, int lane) {
  int row = lane & 15;
  int kb  = (lane >> 4) << 3;
  const _Float16* p = A + (size_t)row * ld + kb;
  v8h lo = *(const v8h*)(p);
  v8h hi = *(const v8h*)(p + 16);
  v16h r;
#pragma unroll
  for (int i = 0; i < 8; ++i) { r[i] = lo[i]; r[i + 8] = hi[i]; }
  return r;
}

// Same but source is f32 (converted on the fly; used for P@V).
static __device__ __forceinline__ v16h frag_a_f32(const float* __restrict__ A,
                                                  int ld, int lane) {
  int row = lane & 15;
  int kb  = (lane >> 4) << 3;
  const float* p = A + (size_t)row * ld + kb;
  v8f lo = *(const v8f*)(p);
  v8f hi = *(const v8f*)(p + 16);
  v16h r;
#pragma unroll
  for (int i = 0; i < 8; ++i) { r[i] = (_Float16)lo[i]; r[i + 8] = (_Float16)hi[i]; }
  return r;
}

// B (32x16 f16) read from Bt where Bt[n][k] is row-major with stride ld:
// lane L holds column N=L%16 of B; halves 0-7 = K kb..kb+7,
// halves 8-15 = K kb+16..kb+23, kb = 8*(L/16).
static __device__ __forceinline__ v16h frag_b_f16(const _Float16* __restrict__ Bt,
                                                  int ld, int lane) {
  int col = lane & 15;
  int kb  = (lane >> 4) << 3;
  const _Float16* p = Bt + (size_t)col * ld + kb;
  v8h lo = *(const v8h*)(p);
  v8h hi = *(const v8h*)(p + 16);
  v16h r;
#pragma unroll
  for (int i = 0; i < 8; ++i) { r[i] = lo[i]; r[i + 8] = hi[i]; }
  return r;
}

// Zero-VGPR-cost pipelining: warm the cacheline of the NEXT k-step fragment.
// Lowers to global_prefetch_b8 (no LOADcnt, no registers held).
static __device__ __forceinline__ void prefetch_a(const _Float16* A, int ld, int lane) {
  __builtin_prefetch(A + (size_t)(lane & 15) * ld + (((lane >> 4) << 3)), 0, 3);
}
static __device__ __forceinline__ void prefetch_a32(const float* A, int ld, int lane) {
  __builtin_prefetch(A + (size_t)(lane & 15) * ld + (((lane >> 4) << 3)), 0, 3);
}
static __device__ __forceinline__ void prefetch_b(const _Float16* Bt, int ld, int lane) {
  __builtin_prefetch(Bt + (size_t)(lane & 15) * ld + (((lane >> 4) << 4)), 0, 3);
}

#define WMMA_F16(a, b, c) \
  __builtin_amdgcn_wmma_f32_16x16x32_f16(false, (a), false, (b), (short)0, (c), false, false)

// ---- f32 -> f16 convert, 4 elements/thread ----
__global__ void cvt_f32_f16_kernel(const float* __restrict__ s,
                                   _Float16* __restrict__ d, int n4) {
  int i = blockIdx.x * blockDim.x + threadIdx.x;
  if (i < n4) {
    v4f v = *(const v4f*)(s + (size_t)i * 4);
    v4h h;
#pragma unroll
    for (int c = 0; c < 4; ++c) h[c] = (_Float16)v[c];
    *(v4h*)(d + (size_t)i * 4) = h;
  }
}

// ---- QKV GEMM: (8192x768) x (768x2304), 32x64 per wave.
//      Epilogue scatters into Qh[bh][n][d], Kh[bh][n][d] (+scale_emb), Vt[bh][d][n].
__global__ void qkv_gemm_kernel(const _Float16* __restrict__ Xh,
                                const _Float16* __restrict__ Wh,   // (2304 x 768) row-major
                                const float* __restrict__ se,      // 768
                                _Float16* __restrict__ Qh,
                                _Float16* __restrict__ Kh,
                                _Float16* __restrict__ Vt) {
  int wid  = (blockIdx.x * blockDim.x + threadIdx.x) >> 5;
  int lane = threadIdx.x & 31;
  const int NT = QKV_N / 64;                  // 36 column chunks
  int m0 = (wid / NT) * 32;
  int n0 = (wid % NT) * 64;
  const _Float16* A0 = Xh + (size_t)m0 * DIMC;
  const _Float16* A1 = A0 + (size_t)16 * DIMC;
  const _Float16* B0 = Wh + (size_t)n0 * DIMC;

  v8f acc[2][4] = {};
#pragma unroll 4
  for (int k0 = 0; k0 < DIMC; k0 += 32) {
    int kn = k0 + 32;
    if (kn < DIMC) {                          // warm next tile, zero reg cost
      prefetch_a(A0 + kn, DIMC, lane);
      prefetch_a(A1 + kn, DIMC, lane);
      prefetch_b(B0 + kn, DIMC, lane);
      prefetch_b(B0 + (size_t)16 * DIMC + kn, DIMC, lane);
      prefetch_b(B0 + (size_t)32 * DIMC + kn, DIMC, lane);
      prefetch_b(B0 + (size_t)48 * DIMC + kn, DIMC, lane);
    }
    v16h a0 = frag_a_f16(A0 + k0, DIMC, lane);
    v16h a1 = frag_a_f16(A1 + k0, DIMC, lane);
    v16h b0 = frag_b_f16(B0 + (size_t)0  * DIMC + k0, DIMC, lane);
    v16h b1 = frag_b_f16(B0 + (size_t)16 * DIMC + k0, DIMC, lane);
    v16h b2 = frag_b_f16(B0 + (size_t)32 * DIMC + k0, DIMC, lane);
    v16h b3 = frag_b_f16(B0 + (size_t)48 * DIMC + k0, DIMC, lane);
    acc[0][0] = WMMA_F16(a0, b0, acc[0][0]);
    acc[1][0] = WMMA_F16(a1, b0, acc[1][0]);
    acc[0][1] = WMMA_F16(a0, b1, acc[0][1]);
    acc[1][1] = WMMA_F16(a1, b1, acc[1][1]);
    acc[0][2] = WMMA_F16(a0, b2, acc[0][2]);
    acc[1][2] = WMMA_F16(a1, b2, acc[1][2]);
    acc[0][3] = WMMA_F16(a0, b3, acc[0][3]);
    acc[1][3] = WMMA_F16(a1, b3, acc[1][3]);
  }

  int col = lane & 15;
  int rb  = (lane >> 4) << 3;
  int bi  = m0 >> 10;                         // batch constant across 32 rows
#pragma unroll
  for (int j = 0; j < 4; ++j) {
    int o0    = n0 + j * 16;                  // uniform column base -> scalar branch
    int which = o0 / DIMC;
    int rem0  = o0 - which * DIMC;
    int h     = rem0 >> 6;
    int d     = (rem0 & 63) + col;
    size_t bh = (size_t)(bi * HEADS + h);
    float sev = (which == 1) ? se[rem0 + col] : 0.0f;
#pragma unroll
    for (int r = 0; r < 2; ++r) {
      int mb = m0 + r * 16 + rb;
#pragma unroll
      for (int v = 0; v < 8; ++v) {
        int n = (mb + v) & (SEQ - 1);
        float f = acc[r][j][v];
        if (which == 0)      Qh[(bh * SEQ + n) * HEAD_DIM + d] = (_Float16)f;
        else if (which == 1) Kh[(bh * SEQ + n) * HEAD_DIM + d] = (_Float16)(f + sev);
        else                 Vt[(bh * HEAD_DIM + d) * SEQ + n] = (_Float16)f;
      }
    }
  }
}

// ---- Logits: S = Q K^T * scale, 32x64 per wave, fully unrolled (K=64) ----
__global__ void logits_gemm_kernel(const _Float16* __restrict__ Qh,
                                   const _Float16* __restrict__ Kh,
                                   float* __restrict__ probs) {
  int wid  = (blockIdx.x * blockDim.x + threadIdx.x) >> 5;
  int lane = threadIdx.x & 31;
  int bh   = wid / 512;                       // 32 m-tiles * 16 n-chunks per head
  int rest = wid - bh * 512;
  int m0   = (rest >> 4) * 32;
  int n0   = (rest & 15) * 64;
  const _Float16* Qb0 = Qh + ((size_t)bh * SEQ + m0) * HEAD_DIM;
  const _Float16* Qb1 = Qb0 + (size_t)16 * HEAD_DIM;
  const _Float16* Kb  = Kh + ((size_t)bh * SEQ + n0) * HEAD_DIM;

  v16h a00 = frag_a_f16(Qb0, HEAD_DIM, lane);
  v16h a01 = frag_a_f16(Qb0 + 32, HEAD_DIM, lane);
  v16h a10 = frag_a_f16(Qb1, HEAD_DIM, lane);
  v16h a11 = frag_a_f16(Qb1 + 32, HEAD_DIM, lane);

  v8f acc[2][4] = {};
#pragma unroll
  for (int j = 0; j < 4; ++j) {
    const _Float16* Bj = Kb + (size_t)(j * 16) * HEAD_DIM;
    v16h b0 = frag_b_f16(Bj, HEAD_DIM, lane);
    v16h b1 = frag_b_f16(Bj + 32, HEAD_DIM, lane);
    acc[0][j] = WMMA_F16(a00, b0, acc[0][j]);
    acc[1][j] = WMMA_F16(a10, b0, acc[1][j]);
    acc[0][j] = WMMA_F16(a01, b1, acc[0][j]);
    acc[1][j] = WMMA_F16(a11, b1, acc[1][j]);
  }

  float* Cb = probs + ((size_t)bh * SEQ + m0) * SEQ + n0;
  int col = lane & 15;
  int rb  = (lane >> 4) << 3;
#pragma unroll
  for (int r = 0; r < 2; ++r)
#pragma unroll
    for (int j = 0; j < 4; ++j)
#pragma unroll
      for (int v = 0; v < 8; ++v)
        Cb[(size_t)(r * 16 + rb + v) * SEQ + j * 16 + col] = acc[r][j][v] * ATT_SCALE;
}

// ---- Row softmax in place: one wave32 per row of 1024 ----
__global__ void softmax_rows_kernel(float* __restrict__ probs) {
  int wid  = (blockIdx.x * blockDim.x + threadIdx.x) >> 5;  // row id
  int lane = threadIdx.x & 31;
  float* p = probs + (size_t)wid * SEQ;
  v4f vals[8];
#pragma unroll
  for (int i = 0; i < 8; ++i)
    vals[i] = *(const v4f*)(p + (size_t)(i * 32 + lane) * 4);
  float m = -3.0e38f;
#pragma unroll
  for (int i = 0; i < 8; ++i)
#pragma unroll
    for (int c = 0; c < 4; ++c) m = fmaxf(m, vals[i][c]);
#pragma unroll
  for (int s = 16; s >= 1; s >>= 1) m = fmaxf(m, __shfl_xor(m, s, 32));
  float sum = 0.0f;
#pragma unroll
  for (int i = 0; i < 8; ++i)
#pragma unroll
    for (int c = 0; c < 4; ++c) {
      float e = __expf(vals[i][c] - m);
      vals[i][c] = e;
      sum += e;
    }
#pragma unroll
  for (int s = 16; s >= 1; s >>= 1) sum += __shfl_xor(sum, s, 32);
  float inv = 1.0f / sum;
#pragma unroll
  for (int i = 0; i < 8; ++i) {
#pragma unroll
    for (int c = 0; c < 4; ++c) vals[i][c] *= inv;
    *(v4f*)(p + (size_t)(i * 32 + lane) * 4) = vals[i];
  }
}

// ---- O = P @ V : 32x64 per wave; writes Oh (8192x768 f16) ----
__global__ void pv_gemm_kernel(const float* __restrict__ probs,
                               const _Float16* __restrict__ Vt,
                               _Float16* __restrict__ Oh) {
  int wid  = (blockIdx.x * blockDim.x + threadIdx.x) >> 5;
  int lane = threadIdx.x & 31;
  int bh   = wid >> 5;                        // 32 m-tiles per head
  int m0   = (wid & 31) * 32;
  const float*    P0 = probs + ((size_t)bh * SEQ + m0) * SEQ;
  const float*    P1 = P0 + (size_t)16 * SEQ;
  const _Float16* Vb = Vt + (size_t)bh * HEAD_DIM * SEQ;

  v8f acc[2][4] = {};
#pragma unroll 4
  for (int k0 = 0; k0 < SEQ; k0 += 32) {
    int kn = k0 + 32;
    if (kn < SEQ) {
      prefetch_a32(P0 + kn, SEQ, lane);
      prefetch_a32(P1 + kn, SEQ, lane);
      prefetch_b(Vb + kn, SEQ, lane);
      prefetch_b(Vb + (size_t)16 * SEQ + kn, SEQ, lane);
      prefetch_b(Vb + (size_t)32 * SEQ + kn, SEQ, lane);
      prefetch_b(Vb + (size_t)48 * SEQ + kn, SEQ, lane);
    }
    v16h a0 = frag_a_f32(P0 + k0, SEQ, lane);
    v16h a1 = frag_a_f32(P1 + k0, SEQ, lane);
    v16h b0 = frag_b_f16(Vb + (size_t)0  * SEQ + k0, SEQ, lane);
    v16h b1 = frag_b_f16(Vb + (size_t)16 * SEQ + k0, SEQ, lane);
    v16h b2 = frag_b_f16(Vb + (size_t)32 * SEQ + k0, SEQ, lane);
    v16h b3 = frag_b_f16(Vb + (size_t)48 * SEQ + k0, SEQ, lane);
    acc[0][0] = WMMA_F16(a0, b0, acc[0][0]);
    acc[1][0] = WMMA_F16(a1, b0, acc[1][0]);
    acc[0][1] = WMMA_F16(a0, b1, acc[0][1]);
    acc[1][1] = WMMA_F16(a1, b1, acc[1][1]);
    acc[0][2] = WMMA_F16(a0, b2, acc[0][2]);
    acc[1][2] = WMMA_F16(a1, b2, acc[1][2]);
    acc[0][3] = WMMA_F16(a0, b3, acc[0][3]);
    acc[1][3] = WMMA_F16(a1, b3, acc[1][3]);
  }

  int bi = bh / HEADS;
  int h  = bh - bi * HEADS;
  _Float16* Ob = Oh + ((size_t)bi * SEQ + m0) * DIMC + h * HEAD_DIM;
  int col = lane & 15;
  int rb  = (lane >> 4) << 3;
#pragma unroll
  for (int r = 0; r < 2; ++r)
#pragma unroll
    for (int j = 0; j < 4; ++j)
#pragma unroll
      for (int v = 0; v < 8; ++v)
        Ob[(size_t)(r * 16 + rb + v) * DIMC + j * 16 + col] = (_Float16)acc[r][j][v];
}

// ---- Output projection: (8192x768) x (768x768) + bias, 32x64 per wave ----
__global__ void proj_gemm_kernel(const _Float16* __restrict__ Oh,
                                 const _Float16* __restrict__ Wph,  // (768 x 768) row-major
                                 const float* __restrict__ bias,
                                 float* __restrict__ out) {
  int wid  = (blockIdx.x * blockDim.x + threadIdx.x) >> 5;
  int lane = threadIdx.x & 31;
  const int NT = DIMC / 64;                   // 12
  int m0 = (wid / NT) * 32;
  int n0 = (wid % NT) * 64;
  const _Float16* A0 = Oh + (size_t)m0 * DIMC;
  const _Float16* A1 = A0 + (size_t)16 * DIMC;
  const _Float16* B0 = Wph + (size_t)n0 * DIMC;

  v8f acc[2][4] = {};
#pragma unroll 4
  for (int k0 = 0; k0 < DIMC; k0 += 32) {
    int kn = k0 + 32;
    if (kn < DIMC) {
      prefetch_a(A0 + kn, DIMC, lane);
      prefetch_a(A1 + kn, DIMC, lane);
      prefetch_b(B0 + kn, DIMC, lane);
      prefetch_b(B0 + (size_t)16 * DIMC + kn, DIMC, lane);
      prefetch_b(B0 + (size_t)32 * DIMC + kn, DIMC, lane);
      prefetch_b(B0 + (size_t)48 * DIMC + kn, DIMC, lane);
    }
    v16h a0 = frag_a_f16(A0 + k0, DIMC, lane);
    v16h a1 = frag_a_f16(A1 + k0, DIMC, lane);
    v16h b0 = frag_b_f16(B0 + (size_t)0  * DIMC + k0, DIMC, lane);
    v16h b1 = frag_b_f16(B0 + (size_t)16 * DIMC + k0, DIMC, lane);
    v16h b2 = frag_b_f16(B0 + (size_t)32 * DIMC + k0, DIMC, lane);
    v16h b3 = frag_b_f16(B0 + (size_t)48 * DIMC + k0, DIMC, lane);
    acc[0][0] = WMMA_F16(a0, b0, acc[0][0]);
    acc[1][0] = WMMA_F16(a1, b0, acc[1][0]);
    acc[0][1] = WMMA_F16(a0, b1, acc[0][1]);
    acc[1][1] = WMMA_F16(a1, b1, acc[1][1]);
    acc[0][2] = WMMA_F16(a0, b2, acc[0][2]);
    acc[1][2] = WMMA_F16(a1, b2, acc[1][2]);
    acc[0][3] = WMMA_F16(a0, b3, acc[0][3]);
    acc[1][3] = WMMA_F16(a1, b3, acc[1][3]);
  }

  int col = lane & 15;
  int rb  = (lane >> 4) << 3;
#pragma unroll
  for (int j = 0; j < 4; ++j) {
    int n = n0 + j * 16 + col;
    float bb = bias[n];
#pragma unroll
    for (int r = 0; r < 2; ++r)
#pragma unroll
      for (int v = 0; v < 8; ++v)
        out[(size_t)(m0 + r * 16 + rb + v) * DIMC + n] = acc[r][j][v] + bb;
  }
}

extern "C" void kernel_launch(void* const* d_in, const int* in_sizes, int n_in,
                              void* d_out, int out_size, void* d_ws, size_t ws_size,
                              hipStream_t stream) {
  (void)in_sizes; (void)n_in; (void)out_size; (void)ws_size;
  const float* x      = (const float*)d_in[0];
  const float* se     = (const float*)d_in[1];
  const float* w_qkv  = (const float*)d_in[2];
  const float* w_proj = (const float*)d_in[3];
  const float* b_proj = (const float*)d_in[4];

  float* out   = (float*)d_out;
  float* probs = out + (size_t)M_TOT * DIMC;   // second tuple output

  char* ws = (char*)d_ws;
  _Float16* Xh  = (_Float16*)ws; ws += (size_t)M_TOT * DIMC * 2;
  _Float16* Wqh = (_Float16*)ws; ws += (size_t)QKV_N * DIMC * 2;
  _Float16* Wph = (_Float16*)ws; ws += (size_t)DIMC * DIMC * 2;
  _Float16* Qh  = (_Float16*)ws; ws += (size_t)NBH * SEQ * HEAD_DIM * 2;
  _Float16* Kh  = (_Float16*)ws; ws += (size_t)NBH * SEQ * HEAD_DIM * 2;
  _Float16* Vt  = (_Float16*)ws; ws += (size_t)NBH * HEAD_DIM * SEQ * 2;
  _Float16* Oh  = (_Float16*)ws; ws += (size_t)M_TOT * DIMC * 2;

  const int TPB = 256;
  {
    int n4 = M_TOT * DIMC / 4;
    cvt_f32_f16_kernel<<<(n4 + TPB - 1) / TPB, TPB, 0, stream>>>(x, Xh, n4);
  }
  {
    int n4 = QKV_N * DIMC / 4;
    cvt_f32_f16_kernel<<<(n4 + TPB - 1) / TPB, TPB, 0, stream>>>(w_qkv, Wqh, n4);
  }
  {
    int n4 = DIMC * DIMC / 4;
    cvt_f32_f16_kernel<<<(n4 + TPB - 1) / TPB, TPB, 0, stream>>>(w_proj, Wph, n4);
  }

  // QKV GEMM: (8192/32)*(2304/64) = 9216 waves
  qkv_gemm_kernel<<<9216 / 8, TPB, 0, stream>>>(Xh, Wqh, se, Qh, Kh, Vt);

  // Logits: 96 heads * 32 m-tiles * 16 n-chunks = 49152 waves
  logits_gemm_kernel<<<49152 / 8, TPB, 0, stream>>>(Qh, Kh, probs);

  // Softmax: 96*1024 rows, one wave each = 98304 waves
  softmax_rows_kernel<<<98304 / 8, TPB, 0, stream>>>(probs);

  // P@V: 96 heads * 32 m-tiles = 3072 waves
  pv_gemm_kernel<<<3072 / 8, TPB, 0, stream>>>(probs, Vt, Oh);

  // Projection: (8192/32)*(768/64) = 3072 waves
  proj_gemm_kernel<<<3072 / 8, TPB, 0, stream>>>(Oh, Wph, b_proj, out);
}
